// TinyLPR_64845416235418
// MI455X (gfx1250) — 16-bit and 8-bit matrix operand order repaired and checked by reading, compiled, not run on hardware
//
#include <hip/hip_runtime.h>
#include <stdint.h>

#define EPS 1e-3f

typedef __attribute__((ext_vector_type(8)))  int      vi8;
typedef __attribute__((ext_vector_type(4)))  int      vi4;
typedef __attribute__((ext_vector_type(2)))  int      vi2;
typedef __attribute__((ext_vector_type(8)))  float    vf8;
typedef __attribute__((ext_vector_type(16))) _Float16 vh16;

struct ConvDesc {
  int B, H, W, Cin;
  int Ho, Wo, Cout;
  int KH, KW, sH, sW;
  int padT, padL;
  int dilH, dilW;
  int K;
  int Mtiles, Ntiles;
};

__global__ __launch_bounds__(128)
void gemm_bin_iu8(const signed char* __restrict__ act,
                  const signed char* __restrict__ bt,
                  float* __restrict__ out,
                  ConvDesc d) {
  const int wave = blockIdx.x * (blockDim.x >> 5) + (threadIdx.x >> 5);
  const int lane = threadIdx.x & 31;
  const int ntPairs = d.Ntiles >> 1;
  const int mt = wave / ntPairs;
  const int np = wave - mt * ntPairs;
  if (mt >= d.Mtiles) return;

  const int half = lane >> 4;
  const int r    = lane & 15;

  const int m  = mt * 16 + r;
  const int wo = m % d.Wo;
  const int t1 = m / d.Wo;
  const int ho = t1 % d.Ho;
  const int b  = t1 / d.Ho;

  const signed char* bt0 = bt + (size_t)(np * 32 + r) * d.K;
  const signed char* bt1 = bt0 + (size_t)16 * d.K;

  vi8 acc0 = {0, 0, 0, 0, 0, 0, 0, 0};
  vi8 acc1 = {0, 0, 0, 0, 0, 0, 0, 0};
  const vi2 z2 = {0, 0};

  const int npat = d.KH * d.KW;
  int ky = 0, kx = 0;
  int k0 = 0;
  for (int p = 0; p < npat; ++p) {
    const int hi = ho * d.sH + ky * d.dilH - d.padT;
    const int wi = wo * d.sW + kx * d.dilW - d.padL;
    const bool valid = (hi >= 0) & (hi < d.H) & (wi >= 0) & (wi < d.W);
    const int hic = valid ? hi : 0;
    const int wic = valid ? wi : 0;
    const signed char* abase =
        act + ((size_t)(b * d.H + hic) * d.W + wic) * (size_t)d.Cin + half * 16;

#pragma unroll 2
    for (int ci0 = 0; ci0 < d.Cin; ci0 += 64) {
      const signed char* ap = abase + ci0;
      vi2 a0 = *(const vi2*)(ap);
      vi2 a1 = *(const vi2*)(ap + 8);
      vi2 a2 = *(const vi2*)(ap + 32);
      vi2 a3 = *(const vi2*)(ap + 40);
      vi8 afrag;
      ((vi2*)&afrag)[0] = valid ? a0 : z2;
      ((vi2*)&afrag)[1] = valid ? a1 : z2;
      ((vi2*)&afrag)[2] = valid ? a2 : z2;
      ((vi2*)&afrag)[3] = valid ? a3 : z2;

      const signed char* bp0 = bt0 + k0 + half * 16;
      const signed char* bp1 = bt1 + k0 + half * 16;
      vi8 bf0, bf1;
      ((vi4*)&bf0)[0] = *(const vi4*)(bp0);
      ((vi4*)&bf0)[1] = *(const vi4*)(bp0 + 32);
      ((vi4*)&bf1)[0] = *(const vi4*)(bp1);
      ((vi4*)&bf1)[1] = *(const vi4*)(bp1 + 32);
      __builtin_prefetch(bp0 + 128, 0, 0);

      acc0 = __builtin_amdgcn_wmma_i32_16x16x64_iu8(true, afrag, true, bf0, acc0,
                                                    false, false);
      acc1 = __builtin_amdgcn_wmma_i32_16x16x64_iu8(true, afrag, true, bf1, acc1,
                                                    false, false);
      k0 += 64;
    }
    if (++kx == d.KW) { kx = 0; ++ky; }
  }

  const int col0 = np * 32 + r;
  const int col1 = col0 + 16;
#pragma unroll
  for (int v = 0; v < 8; ++v) {
    float* orow = out + (size_t)(mt * 16 + v + 8 * half) * d.Cout;
    if (col0 < d.Cout) orow[col0] = (float)acc0[v];
    if (col1 < d.Cout) orow[col1] = (float)acc1[v];
  }
}

__global__ __launch_bounds__(128)
void gemm_conv1_f16(const float* __restrict__ act,
                    const _Float16* __restrict__ bt,
                    float* __restrict__ out, int Mtiles, int Ntiles, int dil) {
  const int wave = blockIdx.x * (blockDim.x >> 5) + (threadIdx.x >> 5);
  const int lane = threadIdx.x & 31;
  const int mt = wave / Ntiles;
  const int nt = wave - mt * Ntiles;
  if (mt >= Mtiles) return;

  const int half = lane >> 4;
  const int r    = lane & 15;
  const int m    = mt * 16 + r;
  const int t    = m & 7;
  const _Float16* btrow = bt + (size_t)(nt * 16 + r) * 384;

  vf8 acc = {0.f, 0.f, 0.f, 0.f, 0.f, 0.f, 0.f, 0.f};
  int k0 = 0;
  for (int kx = 0; kx < 3; ++kx) {
    const int ti = t + (kx - 2) * dil;
    const bool valid = (ti >= 0);
    const float* arow = act + (size_t)(m - t + (valid ? ti : 0)) * 128 + half * 16;
#pragma unroll 2
    for (int ci0 = 0; ci0 < 128; ci0 += 32) {
      vh16 afrag;
#pragma unroll
      for (int c = 0; c < 2; ++c) {
        const float* src = arow + ci0 + c * 8;
#pragma unroll
        for (int j = 0; j < 8; ++j) {
          const float f = src[j];
          afrag[c * 8 + j] = (_Float16)(valid ? f : 0.f);
        }
      }
      vh16 bfrag = *(const vh16*)(btrow + k0 + half * 16);
      acc = __builtin_amdgcn_wmma_f32_16x16x32_f16(false, afrag, false, bfrag,
                                                   (short)0, acc, false, false);
      k0 += 32;
    }
  }
#pragma unroll
  for (int v = 0; v < 8; ++v)
    out[(size_t)(mt * 16 + v + 8 * half) * 128 + (nt * 16 + r)] = acc[v];
}

__global__ void conv0_k(float* __restrict__ y0, const float* __restrict__ x,
                        const float* __restrict__ w0, const float* __restrict__ g_,
                        const float* __restrict__ b_, const float* __restrict__ m_,
                        const float* __restrict__ v_) {
  const int idx = blockIdx.x * 256 + threadIdx.x;
  if (idx >= 128 * 32 * 64 * 64) return;
  const int co = idx & 63;
  const int wp = (idx >> 6) & 63;
  const int hp = (idx >> 12) & 31;
  const int b  = idx >> 17;
  float wv[9];
#pragma unroll
  for (int tw = 0; tw < 9; ++tw)
    wv[tw] = (w0[tw * 64 + co] >= 0.f) ? 1.f : -1.f;
  float mx = -1e30f;
  for (int dy = 0; dy < 2; ++dy)
    for (int dx = 0; dx < 2; ++dx) {
      const int y = 2 * hp + dy, xc = 2 * wp + dx;
      float s = 0.f;
      for (int ky = 0; ky < 3; ++ky) {
        const int iy = y + ky - 1;
        if (iy < 0 || iy >= 64) continue;
        for (int kx = 0; kx < 3; ++kx) {
          const int ix = xc + kx - 1;
          if (ix < 0 || ix >= 128) continue;
          s += wv[ky * 3 + kx] * x[(size_t)(b * 64 + iy) * 128 + ix];
        }
      }
      mx = fmaxf(mx, s);
    }
  y0[idx] = (mx - m_[co]) * rsqrtf(v_[co] + EPS) * g_[co] + b_[co];
}

__global__ void sign8_k(signed char* __restrict__ out, const float* __restrict__ in,
                        const float* __restrict__ bias, int N, int cbits) {
  const int idx = blockIdx.x * 256 + threadIdx.x;
  if (idx >= N) return;
  float v = in[idx];
  if (bias) v += bias[idx & ((1 << cbits) - 1)];
  out[idx] = (v >= 0.f) ? (signed char)1 : (signed char)-1;
}

__global__ void bn_epi_k(float* __restrict__ out, const float* __restrict__ g,
                         const float* __restrict__ m_, const float* __restrict__ v_,
                         const float* __restrict__ b_, int N, int cbits, int do_relu) {
  const int idx = blockIdx.x * 256 + threadIdx.x;
  if (idx >= N) return;
  const int c = idx & ((1 << cbits) - 1);
  float y = (g[idx] - m_[c]) * rsqrtf(v_[c] + EPS) + b_[c];
  if (do_relu) y = fmaxf(y, 0.f);
  out[idx] = y;
}

__global__ void bb_epi_k(float* __restrict__ out, const float* __restrict__ g,
                         const float* __restrict__ prev, const float* __restrict__ m_,
                         const float* __restrict__ v_, const float* __restrict__ b_,
                         const float* __restrict__ bias2, int N, int cbits, int hbits,
                         int wbits, int pool) {
  const int idx = blockIdx.x * 256 + threadIdx.x;
  if (idx >= N) return;
  const int C  = 1 << cbits;
  const int c  = idx & (C - 1);
  const int mm = idx >> cbits;
  float y = (g[idx] - m_[c]) * rsqrtf(v_[c] + EPS) + b_[c];
  if (pool) {
    const int Wo = 1 << wbits;
    const int Ho = 1 << hbits;
    const int wo = mm & (Wo - 1);
    const int t  = mm >> wbits;
    const int ho = t & (Ho - 1);
    const int bb = t >> hbits;
    const int W  = 2 * Wo;
    const float* p0 = prev + (((size_t)(bb * 2 * Ho + 2 * ho) * W + 2 * wo) * C + c);
    y += 0.25f * (p0[0] + p0[C] + p0[(size_t)W * C] + p0[(size_t)W * C + C]);
  }
  out[idx] = fmaxf(y + bias2[c], 0.f);
}

__global__ void relu_k(float* __restrict__ out, const float* __restrict__ in, int N) {
  const int idx = blockIdx.x * 256 + threadIdx.x;
  if (idx < N) out[idx] = fmaxf(in[idx], 0.f);
}

__global__ void add_relu_k(float* __restrict__ out, const float* __restrict__ a,
                           const float* __restrict__ b, int N) {
  const int idx = blockIdx.x * 256 + threadIdx.x;
  if (idx < N) out[idx] = fmaxf(a[idx] + b[idx], 0.f);
}

__global__ void dense_pack_k(signed char* __restrict__ out, const float* __restrict__ y4,
                             int N) {
  const int idx = blockIdx.x * 256 + threadIdx.x;
  if (idx >= N) return;
  const int row = idx >> 11;
  const int f   = idx & 2047;
  const int b   = row >> 3;
  const int t   = row & 7;
  const int h2  = 2 * t + (f >> 10);
  const int w2  = (f >> 7) & 7;
  const int c   = f & 127;
  const int h   = h2 & 7;
  const int w   = w2 + 8 * (h2 >> 3);
  const float v = y4[(((size_t)(b * 8 + h) * 16) + w) * 128 + c];
  out[idx] = (v >= 0.f) ? (signed char)1 : (signed char)-1;
}

__global__ void pack_bt_i8(signed char* __restrict__ bt, const float* __restrict__ w,
                           int K, int Cout, int Npad) {
  const int idx = blockIdx.x * 256 + threadIdx.x;
  if (idx >= Npad * K) return;
  const int n = idx / K;
  const int k = idx - n * K;
  bt[idx] = (n < Cout) ? ((w[(size_t)k * Cout + n] >= 0.f) ? (signed char)1
                                                           : (signed char)-1)
                       : (signed char)0;
}

__global__ void pack_bt_f16(_Float16* __restrict__ bt, const float* __restrict__ w,
                            int K, int Cout) {
  const int idx = blockIdx.x * 256 + threadIdx.x;
  if (idx >= Cout * K) return;
  const int n = idx / K;
  const int k = idx - n * K;
  bt[idx] = (w[(size_t)k * Cout + n] >= 0.f) ? (_Float16)1.f : (_Float16)-1.f;
}

__global__ void softmax85_k(float* __restrict__ out, const float* __restrict__ logits,
                            int rows) {
  const int r = blockIdx.x * 256 + threadIdx.x;
  if (r >= rows) return;
  const float* lp = logits + (size_t)r * 85;
  float mx = lp[0];
  for (int i = 1; i < 85; ++i) mx = fmaxf(mx, lp[i]);
  float s = 0.f;
  float* op = out + (size_t)r * 85;
  for (int i = 0; i < 85; ++i) {
    const float e = expf(lp[i] - mx);
    op[i] = e;
    s += e;
  }
  const float inv = 1.f / s;
  for (int i = 0; i < 85; ++i) op[i] *= inv;
}

static inline ConvDesc mkdesc(int B, int H, int W, int Cin, int Ho, int Wo, int Cout,
                              int KH, int KW, int sH, int sW, int padT, int padL,
                              int dilW, int Npad) {
  ConvDesc d;
  d.B = B; d.H = H; d.W = W; d.Cin = Cin;
  d.Ho = Ho; d.Wo = Wo; d.Cout = Cout;
  d.KH = KH; d.KW = KW; d.sH = sH; d.sW = sW;
  d.padT = padT; d.padL = padL; d.dilH = 1; d.dilW = dilW;
  d.K = KH * KW * Cin;
  d.Mtiles = (B * Ho * Wo) / 16;
  d.Ntiles = Npad / 16;
  return d;
}

static inline void launch_gemm(const signed char* act, const signed char* bt, float* out,
                               const ConvDesc& d, hipStream_t s) {
  const int waves  = d.Mtiles * (d.Ntiles >> 1);
  const int blocks = (waves + 3) / 4;
  gemm_bin_iu8<<<blocks, 128, 0, s>>>(act, bt, out, d);
}

static inline int gb(int n) { return (n + 255) / 256; }

extern "C" void kernel_launch(void* const* d_in, const int* in_sizes, int n_in,
                              void* d_out, int out_size, void* d_ws, size_t ws_size,
                              hipStream_t stream) {
  (void)in_sizes; (void)n_in; (void)out_size; (void)ws_size;
  auto F = [&](int i) { return (const float*)d_in[i]; };

  const float* X    = F(0);
  const float* W0   = F(1);
  const float* BN1G = F(2); const float* BN1B = F(3);
  const float* BN1M = F(4); const float* BN1V = F(5);
  struct BBp { const float *bias1, *w, *bn_m, *bn_v, *bn_b, *bias2; };
  auto getBB = [&](int base) {
    BBp p; p.bias1 = F(base); p.w = F(base + 1); p.bn_m = F(base + 2);
    p.bn_v = F(base + 3); p.bn_b = F(base + 4); p.bias2 = F(base + 5); return p;
  };
  BBp bb1 = getBB(6), bb2 = getBB(12), bb3 = getBB(18);
  const float* LBIAS = F(24); const float* LW = F(25);
  const float* LM = F(26); const float* LV = F(27); const float* LB = F(28);
  const float* WD1 = F(29);
  struct RBp { const float *w_sc, *w1, *bn1_m, *bn1_v, *bn1_b, *w2, *bn2_m, *bn2_v, *bn2_b; };
  auto getRB = [&](int base) {
    RBp p; p.w_sc = F(base); p.w1 = F(base + 1); p.bn1_m = F(base + 2);
    p.bn1_v = F(base + 3); p.bn1_b = F(base + 4); p.w2 = F(base + 5);
    p.bn2_m = F(base + 6); p.bn2_v = F(base + 7); p.bn2_b = F(base + 8); return p;
  };
  RBp rb[3] = { getRB(30), getRB(39), getRB(48) };
  const float* WO = F(57);

  uint8_t* ws = (uint8_t*)d_ws;
  size_t cur = 0;
  auto alloc = [&](size_t bytes) -> void* {
    void* p = ws + cur;
    cur = (cur + bytes + 255) & ~(size_t)255;
    return p;
  };
  signed char* BT1 = (signed char*)alloc(64 * 576);
  signed char* BT2 = (signed char*)alloc(128 * 576);
  signed char* BT3 = (signed char*)alloc(128 * 1152);
  signed char* BT4 = (signed char*)alloc(128 * 768);
  signed char* BTD = (signed char*)alloc(128 * 2048);
  signed char* BTS[3]; _Float16* BTW1[3]; signed char* BTW2[3];
  for (int i = 0; i < 3; ++i) {
    BTS[i]  = (signed char*)alloc(128 * 128);
    BTW1[i] = (_Float16*)alloc(128 * 384 * 2);
    BTW2[i] = (signed char*)alloc(128 * 384);
  }
  signed char* BTO = (signed char*)alloc(96 * 128);
  float* F0 = (float*)alloc((size_t)16777216 * 4);
  float* F1 = (float*)alloc((size_t)8388608 * 4);
  float* F2 = (float*)alloc((size_t)8388608 * 4);
  signed char* I0 = (signed char*)alloc((size_t)16777216);
  float* SM = (float*)alloc((size_t)1048576 * 4);
  float* sc = SM;               float* c1 = SM + 131072;
  float* t1 = SM + 262144;      float* c2 = SM + 393216;
  float* t2 = SM + 524288;      float* lg = SM + 655360;
  signed char* I1 = I0 + (1 << 20);

  pack_bt_i8<<<gb(64 * 576),   256, 0, stream>>>(BT1, bb1.w, 576, 64, 64);
  pack_bt_i8<<<gb(128 * 576),  256, 0, stream>>>(BT2, bb2.w, 576, 128, 128);
  pack_bt_i8<<<gb(128 * 1152), 256, 0, stream>>>(BT3, bb3.w, 1152, 128, 128);
  pack_bt_i8<<<gb(128 * 768),  256, 0, stream>>>(BT4, LW, 768, 128, 128);
  pack_bt_i8<<<gb(128 * 2048), 256, 0, stream>>>(BTD, WD1, 2048, 128, 128);
  for (int i = 0; i < 3; ++i) {
    pack_bt_i8<<<gb(128 * 128), 256, 0, stream>>>(BTS[i],  rb[i].w_sc, 128, 128, 128);
    pack_bt_f16<<<gb(128 * 384), 256, 0, stream>>>(BTW1[i], rb[i].w1, 384, 128);
    pack_bt_i8<<<gb(128 * 384), 256, 0, stream>>>(BTW2[i], rb[i].w2, 384, 128, 128);
  }
  pack_bt_i8<<<gb(96 * 128), 256, 0, stream>>>(BTO, WO, 128, 85, 96);

  conv0_k<<<gb(16777216), 256, 0, stream>>>(F0, X, W0, BN1G, BN1B, BN1M, BN1V);

  sign8_k<<<gb(16777216), 256, 0, stream>>>(I0, F0, bb1.bias1, 16777216, 6);
  ConvDesc d1 = mkdesc(128, 32, 64, 64, 16, 32, 64, 3, 3, 2, 2, 0, 0, 1, 64);
  launch_gemm(I0, BT1, F1, d1, stream);
  bb_epi_k<<<gb(65536 * 64), 256, 0, stream>>>(F2, F1, F0, bb1.bn_m, bb1.bn_v,
                                               bb1.bn_b, bb1.bias2, 65536 * 64,
                                               6, 4, 5, 1);

  sign8_k<<<gb(65536 * 64), 256, 0, stream>>>(I0, F2, bb2.bias1, 65536 * 64, 6);
  ConvDesc d2 = mkdesc(128, 16, 32, 64, 16, 32, 128, 3, 3, 1, 1, 1, 1, 1, 128);
  launch_gemm(I0, BT2, F1, d2, stream);
  bb_epi_k<<<gb(65536 * 128), 256, 0, stream>>>(F0, F1, nullptr, bb2.bn_m, bb2.bn_v,
                                                bb2.bn_b, bb2.bias2, 65536 * 128,
                                                7, 4, 5, 0);

  sign8_k<<<gb(65536 * 128), 256, 0, stream>>>(I0, F0, bb3.bias1, 65536 * 128, 7);
  ConvDesc d3 = mkdesc(128, 16, 32, 128, 8, 16, 128, 3, 3, 2, 2, 0, 0, 1, 128);
  launch_gemm(I0, BT3, F1, d3, stream);
  bb_epi_k<<<gb(16384 * 128), 256, 0, stream>>>(F2, F1, F0, bb3.bn_m, bb3.bn_v,
                                                bb3.bn_b, bb3.bias2, 16384 * 128,
                                                7, 3, 4, 1);

  sign8_k<<<gb(16384 * 128), 256, 0, stream>>>(I0, F2, LBIAS, 16384 * 128, 7);
  ConvDesc d4 = mkdesc(128, 8, 16, 128, 8, 16, 128, 1, 6, 1, 1, 0, 2, 1, 128);
  launch_gemm(I0, BT4, F1, d4, stream);
  bn_epi_k<<<gb(16384 * 128), 256, 0, stream>>>(F0, F1, LM, LV, LB, 16384 * 128, 7, 0);

  dense_pack_k<<<gb(1024 * 2048), 256, 0, stream>>>(I0, F0, 1024 * 2048);
  ConvDesc dd = mkdesc(1024, 1, 1, 2048, 1, 1, 128, 1, 1, 1, 1, 0, 0, 1, 128);
  launch_gemm(I0, BTD, F1, dd, stream);
  relu_k<<<gb(131072), 256, 0, stream>>>(F2, F1, 131072);

  float* h  = F2;
  float* hn = F0;
  const int dils[3] = {1, 2, 4};
  for (int i = 0; i < 3; ++i) {
    const int dil = dils[i];
    sign8_k<<<gb(131072), 256, 0, stream>>>(I0, h, nullptr, 131072, 7);
    ConvDesc ds = mkdesc(128, 1, 8, 128, 1, 8, 128, 1, 1, 1, 1, 0, 0, 1, 128);
    launch_gemm(I0, BTS[i], sc, ds, stream);
    gemm_conv1_f16<<<(64 * 8 + 3) / 4, 128, 0, stream>>>(h, BTW1[i], c1, 64, 8, dil);
    bn_epi_k<<<gb(131072), 256, 0, stream>>>(t1, c1, rb[i].bn1_m, rb[i].bn1_v,
                                             rb[i].bn1_b, 131072, 7, 1);
    sign8_k<<<gb(131072), 256, 0, stream>>>(I1, t1, nullptr, 131072, 7);
    ConvDesc dc2 = mkdesc(128, 1, 8, 128, 1, 8, 128, 1, 3, 1, 1, 0, 2 * dil, dil, 128);
    launch_gemm(I1, BTW2[i], c2, dc2, stream);
    bn_epi_k<<<gb(131072), 256, 0, stream>>>(t2, c2, rb[i].bn2_m, rb[i].bn2_v,
                                             rb[i].bn2_b, 131072, 7, 1);
    add_relu_k<<<gb(131072), 256, 0, stream>>>(hn, t2, sc, 131072);
    float* tmp = h; h = hn; hn = tmp;
  }

  sign8_k<<<gb(131072), 256, 0, stream>>>(I0, h, nullptr, 131072, 7);
  ConvDesc doh = mkdesc(1024, 1, 1, 128, 1, 1, 85, 1, 1, 1, 1, 0, 0, 1, 96);
  launch_gemm(I0, BTO, lg, doh, stream);
  softmax85_k<<<gb(1024), 256, 0, stream>>>((float*)d_out, lg, 1024);
}
